// GRUD_88330297410067
// MI455X (gfx1250) — matrix operand order repaired and checked by reading, not compiled, not run on hardware
//
#include <hip/hip_runtime.h>
#include <cstdint>
#include <cstddef>

// ---------------------------------------------------------------------------
// GRU-D fused persistent kernel for gfx1250 (MI455X), bf16 WMMA 16x16x32.
// B=128, T=256, D=512, H=1024.
// ---------------------------------------------------------------------------

typedef __bf16 bf16_t;
typedef __attribute__((ext_vector_type(16))) __bf16 v16bf;
typedef __attribute__((ext_vector_type(8)))  float  v8f;

#define B_   128
#define T_   256
#define D_   512
#define H_   1024
#define KTOT 1536                 // D + H
#define MT   32                   // batch rows per workgroup (2 M-tiles of 16)
#define NBG  (B_ / MT)            // 4 batch groups
#define NJG  (H_ / 32)            // 32 column groups (32 h-cols each)
#define NWG  (NBG * NJG)          // 128 workgroups, 64 threads each (2 waves)

// round-to-nearest-even f32 -> bf16 (bit pattern)
__device__ __forceinline__ unsigned short f2bf(float f) {
    unsigned int u = __float_as_uint(f);
    u += 0x7FFFu + ((u >> 16) & 1u);
    return (unsigned short)(u >> 16);
}

__device__ __forceinline__ float sigmoidf_(float x) {
    return 1.0f / (1.0f + __expf(-x));
}

union Frag16 {                    // 16 bf16 elements = one WMMA A/B fragment
    v16bf v;
    int4  q[2];
};

// ---------------------------------------------------------------------------
// Prep kernels: repack weights into Wcat bf16 (3H x (D+H)), convert x to bf16.
// ---------------------------------------------------------------------------
__global__ void prep_w_kernel(const float* __restrict__ wih,
                              const float* __restrict__ whh,
                              unsigned short* __restrict__ wcat) {
    int idx = blockIdx.x * 256 + threadIdx.x;        // 3072*1536 total
    int n = idx / KTOT;
    int c = idx - n * KTOT;
    float v = (c < D_) ? wih[(size_t)n * D_ + c]
                       : whh[(size_t)n * H_ + (c - D_)];
    wcat[idx] = f2bf(v);
}

__global__ void prep_x_kernel(const float* __restrict__ x,
                              unsigned short* __restrict__ xbf) {
    size_t idx = (size_t)blockIdx.x * 256 + threadIdx.x;  // B*T*D total
    xbf[idx] = f2bf(x[idx]);
}

// ---------------------------------------------------------------------------
// Persistent recurrent kernel. 128 WGs x 64 threads. One grid barrier / step.
// ---------------------------------------------------------------------------
__global__ __launch_bounds__(64)
void grud_main_kernel(const float* __restrict__ times,
                      const int*   __restrict__ length,
                      const float* __restrict__ b_ih,
                      const float* __restrict__ b_hh,
                      const float* __restrict__ w_decay,
                      const float* __restrict__ b_decay,
                      const unsigned short* __restrict__ wcat,
                      const unsigned short* __restrict__ xbf,
                      unsigned short* __restrict__ hbf,   // 2 * B*H bf16 (double buffer)
                      int* __restrict__ bar_cnt,
                      float* __restrict__ out_seq,        // (B,T,H)
                      float* __restrict__ out_end)        // (B,H)
{
    const int wg    = blockIdx.x;
    const int bg    = wg % NBG;            // batch group
    const int jg    = wg / NBG;            // column group
    const int tid   = threadIdx.x;
    const int wv    = tid >> 5;            // wave 0/1 within WG
    const int lane  = tid & 31;
    const int lhalf = lane & 15;
    const bool hi   = (lane >= 16);

    const int bbase = bg * MT;                      // 32 batch rows
    const int j     = jg * 32 + wv * 16 + lhalf;    // h-column owned by this lane (C/B col)

    // per-column constants
    const float wd  = w_decay[j];
    const float bd  = b_decay[j];
    const float br  = b_ih[j]          + b_hh[j];
    const float bi  = b_ih[H_ + j]     + b_hh[H_ + j];
    const float bnx = b_ih[2 * H_ + j];
    const float bnh = b_hh[2 * H_ + j];

    // Wcat rows for this lane's B-fragment column (gate r / i / n)
    const unsigned short* wR = wcat + (size_t)(j)          * KTOT;
    const unsigned short* wI = wcat + (size_t)(H_ + j)     * KTOT;
    const unsigned short* wN = wcat + (size_t)(2 * H_ + j) * KTOT;

    // C-fragment rows: m-tile0 rows bbase+(hi?8:0)+r, m-tile1 those +16
    const int crow0 = bbase + (hi ? 8 : 0);
    int len0[8], len1[8];
#pragma unroll
    for (int r = 0; r < 8; ++r) {
        int L0 = length[crow0 + r] - 1;
        int L1 = length[crow0 + 16 + r] - 1;
        len0[r] = L0 < 0 ? 0 : (L0 > T_ - 1 ? T_ - 1 : L0);
        len1[r] = L1 < 0 ? 0 : (L1 > T_ - 1 ? T_ - 1 : L1);
    }

    // previous hidden state (this WG's own columns), h0 = 0
    float hprev0[8], hprev1[8];
#pragma unroll
    for (int r = 0; r < 8; ++r) { hprev0[r] = 0.f; hprev1[r] = 0.f; }

    // A-fragment row bases (row = lane%16, K halves split by lane-half)
    const int kaddA = hi ? 16 : 0;
    const size_t arow0 = (size_t)(bbase + lhalf);
    const size_t arow1 = (size_t)(bbase + 16 + lhalf);

    for (int t = 0; t < T_; ++t) {
        unsigned short* hcur = hbf + (size_t)(t & 1) * (B_ * H_);

        // ---------- Phase A: decay own columns, publish bf16 h_dec ----------
        float hdec0[8], hdec1[8];
#pragma unroll
        for (int r = 0; r < 8; ++r) {
            int b0 = crow0 + r;
            float e0 = times[(size_t)b0 * T_ + t] * wd + bd;
            e0 = fminf(fmaxf(e0, 0.f), 1000.f);
            hdec0[r] = hprev0[r] * __expf(-e0);
            hcur[(size_t)b0 * H_ + j] = f2bf(hdec0[r]);

            int b1 = b0 + 16;
            float e1 = times[(size_t)b1 * T_ + t] * wd + bd;
            e1 = fminf(fmaxf(e1, 0.f), 1000.f);
            hdec1[r] = hprev1[r] * __expf(-e1);
            hcur[(size_t)b1 * H_ + j] = f2bf(hdec1[r]);
        }

        // ---------- grid barrier (monotonic counter) ----------
        __syncthreads();
        if (tid == 0) {
            __threadfence();   // make hcur stores device-visible
            __hip_atomic_fetch_add(bar_cnt, 1, __ATOMIC_ACQ_REL,
                                   __HIP_MEMORY_SCOPE_AGENT);
            const int target = (t + 1) * NWG;
            while (__hip_atomic_load(bar_cnt, __ATOMIC_ACQUIRE,
                                     __HIP_MEMORY_SCOPE_AGENT) < target) {
                __builtin_amdgcn_s_sleep(2);
            }
        }
        __syncthreads();
        __threadfence();       // acquire: invalidate local caches before reading hcur

        // ---------- Phase B: [x_t | h_dec] @ Wcat^T via WMMA ----------
        v8f accR0 = {0,0,0,0,0,0,0,0}, accI0 = {0,0,0,0,0,0,0,0};
        v8f accXN0 = {0,0,0,0,0,0,0,0}, accHN0 = {0,0,0,0,0,0,0,0};
        v8f accR1 = {0,0,0,0,0,0,0,0}, accI1 = {0,0,0,0,0,0,0,0};
        v8f accXN1 = {0,0,0,0,0,0,0,0}, accHN1 = {0,0,0,0,0,0,0,0};

        const unsigned short* xr0 = xbf + (arow0 * T_ + t) * (size_t)D_;
        const unsigned short* xr1 = xbf + (arow1 * T_ + t) * (size_t)D_;
        const unsigned short* hr0 = hcur + arow0 * H_;
        const unsigned short* hr1 = hcur + arow1 * H_;
        const int kaddB = hi ? 16 : 0;

        // K = 0..511 : x part (feeds accR/accI/accXN)
#pragma unroll 2
        for (int kb = 0; kb < D_; kb += 32) {
            Frag16 a0, a1, fR, fI, fN;
            const int ka = kb + kaddA;
            a0.q[0] = *(const int4*)(xr0 + ka);
            a0.q[1] = *(const int4*)(xr0 + ka + 8);
            a1.q[0] = *(const int4*)(xr1 + ka);
            a1.q[1] = *(const int4*)(xr1 + ka + 8);
            const int kB = kb + kaddB;
            fR.q[0] = *(const int4*)(wR + kB); fR.q[1] = *(const int4*)(wR + kB + 8);
            fI.q[0] = *(const int4*)(wI + kB); fI.q[1] = *(const int4*)(wI + kB + 8);
            fN.q[0] = *(const int4*)(wN + kB); fN.q[1] = *(const int4*)(wN + kB + 8);

            accR0  = __builtin_amdgcn_wmma_f32_16x16x32_bf16(false, a0.v, false, fR.v, (short)0, accR0,  false, false);
            accI0  = __builtin_amdgcn_wmma_f32_16x16x32_bf16(false, a0.v, false, fI.v, (short)0, accI0,  false, false);
            accXN0 = __builtin_amdgcn_wmma_f32_16x16x32_bf16(false, a0.v, false, fN.v, (short)0, accXN0, false, false);
            accR1  = __builtin_amdgcn_wmma_f32_16x16x32_bf16(false, a1.v, false, fR.v, (short)0, accR1,  false, false);
            accI1  = __builtin_amdgcn_wmma_f32_16x16x32_bf16(false, a1.v, false, fI.v, (short)0, accI1,  false, false);
            accXN1 = __builtin_amdgcn_wmma_f32_16x16x32_bf16(false, a1.v, false, fN.v, (short)0, accXN1, false, false);
        }

        // K = 512..1535 : h part (feeds accR/accI/accHN)
#pragma unroll 2
        for (int kb = 0; kb < H_; kb += 32) {
            Frag16 a0, a1, fR, fI, fN;
            const int ka = kb + kaddA;
            a0.q[0] = *(const int4*)(hr0 + ka);
            a0.q[1] = *(const int4*)(hr0 + ka + 8);
            a1.q[0] = *(const int4*)(hr1 + ka);
            a1.q[1] = *(const int4*)(hr1 + ka + 8);
            const int kB = D_ + kb + kaddB;
            fR.q[0] = *(const int4*)(wR + kB); fR.q[1] = *(const int4*)(wR + kB + 8);
            fI.q[0] = *(const int4*)(wI + kB); fI.q[1] = *(const int4*)(wI + kB + 8);
            fN.q[0] = *(const int4*)(wN + kB); fN.q[1] = *(const int4*)(wN + kB + 8);

            accR0  = __builtin_amdgcn_wmma_f32_16x16x32_bf16(false, a0.v, false, fR.v, (short)0, accR0,  false, false);
            accI0  = __builtin_amdgcn_wmma_f32_16x16x32_bf16(false, a0.v, false, fI.v, (short)0, accI0,  false, false);
            accHN0 = __builtin_amdgcn_wmma_f32_16x16x32_bf16(false, a0.v, false, fN.v, (short)0, accHN0, false, false);
            accR1  = __builtin_amdgcn_wmma_f32_16x16x32_bf16(false, a1.v, false, fR.v, (short)0, accR1,  false, false);
            accI1  = __builtin_amdgcn_wmma_f32_16x16x32_bf16(false, a1.v, false, fI.v, (short)0, accI1,  false, false);
            accHN1 = __builtin_amdgcn_wmma_f32_16x16x32_bf16(false, a1.v, false, fN.v, (short)0, accHN1, false, false);
        }

        // ---------- gates + output ----------
#pragma unroll
        for (int r = 0; r < 8; ++r) {
            {
                float rg = sigmoidf_(accR0[r] + br);
                float ig = sigmoidf_(accI0[r] + bi);
                float ng = tanhf(accXN0[r] + bnx + rg * (accHN0[r] + bnh));
                float hy = ng + ig * (hdec0[r] - ng);
                hprev0[r] = hy;
                int b0 = crow0 + r;
                out_seq[((size_t)b0 * T_ + t) * H_ + j] = hy;
                if (t == len0[r]) out_end[(size_t)b0 * H_ + j] = hy;
            }
            {
                float rg = sigmoidf_(accR1[r] + br);
                float ig = sigmoidf_(accI1[r] + bi);
                float ng = tanhf(accXN1[r] + bnx + rg * (accHN1[r] + bnh));
                float hy = ng + ig * (hdec1[r] - ng);
                hprev1[r] = hy;
                int b1 = crow0 + 16 + r;
                out_seq[((size_t)b1 * T_ + t) * H_ + j] = hy;
                if (t == len1[r]) out_end[(size_t)b1 * H_ + j] = hy;
            }
        }
    }
}

// ---------------------------------------------------------------------------
extern "C" void kernel_launch(void* const* d_in, const int* in_sizes, int n_in,
                              void* d_out, int out_size, void* d_ws, size_t ws_size,
                              hipStream_t stream) {
    const float* xin     = (const float*)d_in[0];
    const float* times   = (const float*)d_in[1];
    /* d_in[2] = mask, unused */
    const int*   length  = (const int*)d_in[3];
    const float* w_ih    = (const float*)d_in[4];
    const float* w_hh    = (const float*)d_in[5];
    const float* b_ih    = (const float*)d_in[6];
    const float* b_hh    = (const float*)d_in[7];
    const float* w_decay = (const float*)d_in[8];
    const float* b_decay = (const float*)d_in[9];

    // workspace layout
    char* ws = (char*)d_ws;
    int* bar_cnt           = (int*)ws;                              // 256 B
    unsigned short* wcat   = (unsigned short*)(ws + 256);           // 3H*KTOT bf16
    unsigned short* xbf    = wcat + (size_t)3 * H_ * KTOT;          // B*T*D bf16
    unsigned short* hbf    = xbf + (size_t)B_ * T_ * D_;            // 2*B*H bf16
    const size_t ws_needed = 256
        + (size_t)3 * H_ * KTOT * 2
        + (size_t)B_ * T_ * D_ * 2
        + (size_t)2 * B_ * H_ * 2;
    if (ws_size < ws_needed) return;   // insufficient scratch: nothing we can do

    float* out_seq = (float*)d_out;
    float* out_end = out_seq + (size_t)B_ * T_ * H_;

    hipMemsetAsync(bar_cnt, 0, 256, stream);
    prep_w_kernel<<<(3 * H_ * KTOT) / 256, 256, 0, stream>>>(w_ih, w_hh, wcat);
    prep_x_kernel<<<(B_ * T_ * D_) / 256, 256, 0, stream>>>(xin, xbf);
    grud_main_kernel<<<NWG, 64, 0, stream>>>(times, length, b_ih, b_hh,
                                             w_decay, b_decay, wcat, xbf, hbf,
                                             bar_cnt, out_seq, out_end);
}
